// SchNetUpdate_10943576671045
// MI455X (gfx1250) — matrix operand order repaired and checked by reading, not compiled, not run on hardware
//
#include <hip/hip_runtime.h>

#define NUM_RBF 300
#define KPAD    320
#define H       64
#define MB      32
#define ATOM    64
#define GAMMA   10.0f
#define RES     0.1f
#define LOG2E   1.4426950408889634f

typedef __attribute__((ext_vector_type(16))) _Float16 v16h;
typedef __attribute__((ext_vector_type(8)))  _Float16 v8h;
typedef __attribute__((ext_vector_type(8)))  float    v8f;

__device__ __forceinline__ float softplus_f(float x) {
  // numerically stable softplus: max(x,0) + log(1 + exp(-|x|))
  float e = __expf(-fabsf(x));
  return fmaxf(x, 0.0f) + __logf(1.0f + e);
}

// ---------------------------------------------------------------------------
// Kernel 0: convert Wd1 (padded to K=320) and Wd2 to f16 "B-matrix" storage.
// wd1h[n][k] = Wd1[n][k] (k<300 else 0), wd2h[n][k] = Wd2[n][k]; row-major.
// ---------------------------------------------------------------------------
__global__ void prep_weights_kernel(const float* __restrict__ Wd1,
                                    const float* __restrict__ Wd2,
                                    _Float16* __restrict__ wd1h,
                                    _Float16* __restrict__ wd2h) {
  int idx = blockIdx.x * blockDim.x + threadIdx.x;
  if (idx < H * KPAD) {
    int n = idx / KPAD, k = idx % KPAD;
    wd1h[idx] = (k < NUM_RBF) ? (_Float16)Wd1[n * NUM_RBF + k] : (_Float16)0.0f;
  } else {
    int j = idx - H * KPAD;
    if (j < H * H) wd2h[j] = (_Float16)Wd2[j];
  }
}

// ---------------------------------------------------------------------------
// Kernel 1: v1[m,i,c] = sum_c' x[m,i,c'] * W1[c,c'] + b1[c]   (tiny: 8 MFLOP)
// ---------------------------------------------------------------------------
__global__ void linear1_kernel(const float* __restrict__ x,
                               const float* __restrict__ W1,
                               const float* __restrict__ b1,
                               float* __restrict__ v1) {
  int idx = blockIdx.x * blockDim.x + threadIdx.x;  // over [MB*ATOM][H]
  int c = idx & (H - 1);
  int row = idx >> 6;
  const float* xr = x + row * H;
  const float* wr = W1 + c * H;
  float s = b1[c];
#pragma unroll 8
  for (int k = 0; k < H; ++k) s = fmaf(xr[k], wr[k], s);
  v1[idx] = s;
}

// ---------------------------------------------------------------------------
// Kernel 2: fused CFConv. Block = (jt, m, ih): 16 target atoms j of batch m,
// source-atom half ih (i in [ih*32, ih*32+32)). 8 waves; wave w handles
// i = ih*32 + w, +8, +16, +24.
//   layer1: A = rbf(dist) generated in-registers (v_exp), B = Wd1^T (LDS),
//           10 K-steps of v_wmma_f32_16x16x32_f16 per N-tile (4 N-tiles).
//   layer2: A = softplus(f1) via per-wave LDS bounce, B = Wd2^T, 2 K-steps.
//   reduce: acc[j,c] += v1[i,c] * softplus(f2[j,c])   (scalar per lane).
// Cross-wave i-reduction through LDS; partial sums -> v2part[ih][m,j,c]
// (two buffers, summed deterministically in post_kernel — no float atomics).
// ---------------------------------------------------------------------------
__global__ __launch_bounds__(256)
void cfconv_wmma_kernel(const float* __restrict__ dist,
                        const float* __restrict__ v1,
                        const _Float16* __restrict__ wd1h,
                        const _Float16* __restrict__ wd2h,
                        const float* __restrict__ bd1,
                        const float* __restrict__ bd2,
                        float* __restrict__ v2part) {
  __shared__ __attribute__((aligned(16))) _Float16 ldsB1[H * KPAD];   // 40 KB
  __shared__ __attribute__((aligned(16))) _Float16 ldsB2[H * H];      //  8 KB
  __shared__ __attribute__((aligned(16))) float    ldsV1[ATOM * H];   // 16 KB
  __shared__ __attribute__((aligned(16))) _Float16 ldsF1[8][16 * H];  // 16 KB
  __shared__ __attribute__((aligned(16))) float    ldsAcc[8][16 * H]; // 32 KB

  const int tid = threadIdx.x;
  const int jt  = blockIdx.x;          // 0..3, j range = jt*16 .. jt*16+15
  const int m   = blockIdx.y;
  const int ih  = blockIdx.z;          // 0..1, i range = ih*32 .. ih*32+31
  const int w   = tid >> 5;            // wave id 0..7
  const int lane = tid & 31;
  const int ln  = lane & 15;           // N / row-in-tile index
  const int hi  = lane >> 4;           // lane half

  // cooperative stage of weights and v1[m,:,:] -- 16B (b128) granularity
  {
    const uint4* s1 = (const uint4*)wd1h;                 // 2560 vec4
    const uint4* s2 = (const uint4*)wd2h;                 //  512 vec4
    const uint4* s3 = (const uint4*)(v1 + m * ATOM * H);  // 1024 vec4
    uint4* d1 = (uint4*)ldsB1;
    uint4* d2 = (uint4*)ldsB2;
    uint4* d3 = (uint4*)ldsV1;
    for (int idx = tid; idx < (H * KPAD) / 8; idx += 256) d1[idx] = s1[idx];
    for (int idx = tid; idx < (H * H) / 8;    idx += 256) d2[idx] = s2[idx];
    for (int idx = tid; idx < (ATOM * H) / 4; idx += 256) d3[idx] = s3[idx];
  }
  __syncthreads();

  // biases for this lane's 4 output columns (C-layout: N = ln + 16*nt)
  float bd1c[4], bd2c[4];
#pragma unroll
  for (int nt = 0; nt < 4; ++nt) {
    bd1c[nt] = bd1[nt * 16 + ln];
    bd2c[nt] = bd2[nt * 16 + ln];
  }

  float acc[4][8];
#pragma unroll
  for (int nt = 0; nt < 4; ++nt)
#pragma unroll
    for (int g = 0; g < 8; ++g) acc[nt][g] = 0.0f;

  const int jglob = jt * 16 + ln;      // A-layout: row M = lane%16 = local j

  for (int it = 0; it < 4; ++it) {
    const int i = ih * 32 + it * 8 + w;  // source atom handled by this wave
    const float d = dist[(m * ATOM + i) * ATOM + jglob];
    // prefetch next iteration's distance row (global_prefetch_b8)
    __builtin_prefetch(&dist[(m * ATOM + ((i + 8) & (ATOM - 1))) * ATOM + jglob], 0, 0);

    // exp(-g*(d - ck)^2) == exp2(NEGGL2 * t^2), t = d - ck ; ck, NEGGL2 literals
    const float NEGGL2 = -GAMMA * LOG2E;

    // ---------------- layer 1: rbf -> f1 ----------------
    v8f c1[4];
#pragma unroll
    for (int nt = 0; nt < 4; ++nt) c1[nt] = (v8f){0,0,0,0,0,0,0,0};

#pragma unroll
    for (int kt = 0; kt < KPAD / 32; ++kt) {
      // build A fragment: rbf values for this row (j), K-slice kt*32..+31
      v16h a;
#pragma unroll
      for (int h = 0; h < 16; ++h) {
        int kl = h + 16 * hi;   // contiguous k = 16*hi + h (matches B)
        int k  = kt * 32 + kl;
        float t = d - RES * (float)k;
        float r = __builtin_amdgcn_exp2f(NEGGL2 * t * t);  // v_exp_f32
        a[h] = (k < NUM_RBF) ? (_Float16)r : (_Float16)0.0f;
      }
#pragma unroll
      for (int nt = 0; nt < 4; ++nt) {
        const v16h b = *(const v16h*)&ldsB1[(nt * 16 + ln) * KPAD + kt * 32 + 16 * hi];
        c1[nt] = __builtin_amdgcn_wmma_f32_16x16x32_f16(
            false, a, false, b, (short)0, c1[nt], false, false);
      }
    }

    // softplus(f1 + bd1) -> per-wave LDS bounce (C layout -> memory row-major)
#pragma unroll
    for (int nt = 0; nt < 4; ++nt)
#pragma unroll
      for (int g = 0; g < 8; ++g) {
        int row = g + 8 * hi;
        ldsF1[w][row * H + nt * 16 + ln] = (_Float16)softplus_f(c1[nt][g] + bd1c[nt]);
      }
    // wave-private LDS RAW fence (all DS stores of this wave complete)
    asm volatile("s_wait_dscnt 0" ::: "memory");

    // ---------------- layer 2: f1 -> f2 ----------------
    v8f c2[4];
#pragma unroll
    for (int nt = 0; nt < 4; ++nt) c2[nt] = (v8f){0,0,0,0,0,0,0,0};

#pragma unroll
    for (int kt = 0; kt < 2; ++kt) {
      const v8h p = *(const v8h*)&ldsF1[w][ln * H + kt * 32 + 16 * hi];
      const v8h q = *(const v8h*)&ldsF1[w][ln * H + kt * 32 + 16 * hi + 8];
      v16h a2;
#pragma unroll
      for (int h = 0; h < 8; ++h) { a2[h] = p[h]; a2[h + 8] = q[h]; }
#pragma unroll
      for (int nt = 0; nt < 4; ++nt) {
        const v16h b = *(const v16h*)&ldsB2[(nt * 16 + ln) * H + kt * 32 + 16 * hi];
        c2[nt] = __builtin_amdgcn_wmma_f32_16x16x32_f16(
            false, a2, false, b, (short)0, c2[nt], false, false);
      }
    }

    // ---------------- reduction over i ----------------
    // out[j,c] += v1[i,c] * softplus(f2[j,c]); v1 is scalar per (lane, nt).
#pragma unroll
    for (int nt = 0; nt < 4; ++nt) {
      const float v1s = ldsV1[i * H + nt * 16 + ln];
#pragma unroll
      for (int g = 0; g < 8; ++g)
        acc[nt][g] = fmaf(v1s, softplus_f(c2[nt][g] + bd2c[nt]), acc[nt][g]);
    }
  }

  // spill per-wave partials, reduce across the 8 waves (= i mod 8 classes)
#pragma unroll
  for (int nt = 0; nt < 4; ++nt)
#pragma unroll
    for (int g = 0; g < 8; ++g)
      ldsAcc[w][(g + 8 * hi) * H + nt * 16 + ln] = acc[nt][g];
  __syncthreads();

  float* v2p = v2part + (size_t)ih * MB * ATOM * H;
  for (int o = tid; o < 16 * H; o += 256) {
    float s = 0.0f;
#pragma unroll
    for (int ww = 0; ww < 8; ++ww) s += ldsAcc[ww][o];
    int j = o >> 6, c = o & (H - 1);
    v2p[(m * ATOM + jt * 16 + j) * H + c] = s;
  }
}

// ---------------------------------------------------------------------------
// Kernel 3: out = x + W3·softplus(W2·(v2p0+v2p1) + b2) + b3   (tiny)
// one block per (m, j), 64 threads = one output channel each
// ---------------------------------------------------------------------------
__global__ void post_kernel(const float* __restrict__ x,
                            const float* __restrict__ v2part,
                            const float* __restrict__ W2,
                            const float* __restrict__ b2,
                            const float* __restrict__ W3,
                            const float* __restrict__ b3,
                            float* __restrict__ out) {
  __shared__ float sv[H];
  __shared__ float st[H];
  const int row = blockIdx.x;          // m*ATOM + j
  const int o = threadIdx.x;
  sv[o] = v2part[row * H + o] + v2part[(size_t)MB * ATOM * H + row * H + o];
  __syncthreads();
  {
    const float* wr = W2 + o * H;
    float s = b2[o];
#pragma unroll 8
    for (int k = 0; k < H; ++k) s = fmaf(wr[k], sv[k], s);
    st[o] = softplus_f(s);
  }
  __syncthreads();
  {
    const float* wr = W3 + o * H;
    float u = b3[o];
#pragma unroll 8
    for (int k = 0; k < H; ++k) u = fmaf(wr[k], st[k], u);
    out[row * H + o] = x[row * H + o] + u;
  }
}

// ---------------------------------------------------------------------------
extern "C" void kernel_launch(void* const* d_in, const int* in_sizes, int n_in,
                              void* d_out, int out_size, void* d_ws, size_t ws_size,
                              hipStream_t stream) {
  const float* x    = (const float*)d_in[0];
  const float* dist = (const float*)d_in[1];
  const float* W1   = (const float*)d_in[2];
  const float* b1   = (const float*)d_in[3];
  const float* W2   = (const float*)d_in[4];
  const float* b2   = (const float*)d_in[5];
  const float* W3   = (const float*)d_in[6];
  const float* b3   = (const float*)d_in[7];
  const float* Wd1  = (const float*)d_in[8];
  const float* bd1  = (const float*)d_in[9];
  const float* Wd2  = (const float*)d_in[10];
  const float* bd2  = (const float*)d_in[11];
  float* out = (float*)d_out;

  // workspace layout (16B-aligned sections)
  char* ws = (char*)d_ws;
  const size_t NV = (size_t)MB * ATOM * H;        // 131072 elements
  float*    v1     = (float*)(ws);                // 512 KB
  float*    v2part = (float*)(ws + NV * 4);       // 2 x 512 KB partials
  _Float16* wd1h   = (_Float16*)(ws + 3 * NV * 4);                 // 40 KB
  _Float16* wd2h   = (_Float16*)(ws + 3 * NV * 4 + H * KPAD * 2);  //  8 KB

  // 0) f16 weight prep
  {
    int n = H * KPAD + H * H;
    prep_weights_kernel<<<(n + 255) / 256, 256, 0, stream>>>(Wd1, Wd2, wd1h, wd2h);
  }
  // 1) per-atom linear1
  {
    int n = MB * ATOM * H;
    linear1_kernel<<<n / 256, 256, 0, stream>>>(x, W1, b1, v1);
  }
  // 2) fused CFConv (WMMA), i-split over blockIdx.z for occupancy
  {
    dim3 grid(ATOM / 16, MB, 2);
    cfconv_wmma_kernel<<<grid, 256, 0, stream>>>(dist, v1, wd1h, wd2h, bd1, bd2, v2part);
  }
  // 3) output MLP + residual (sums the two i-half partials deterministically)
  post_kernel<<<MB * ATOM, H, 0, stream>>>(x, v2part, W2, b2, W3, b3, out);
}
